// TreeAgent_46145128628802
// MI455X (gfx1250) — matrix operand order repaired and checked by reading, not compiled, not run on hardware
//
#include <hip/hip_runtime.h>
#include <math.h>

// Problem constants (from reference)
#define EXP   64
#define NLEAF 1024
#define KOUT  10
#define BATCH 256
#define DIN   256
#define HID   512

typedef __attribute__((ext_vector_type(16))) __bf16 v16bf;
typedef __attribute__((ext_vector_type(8)))  float  v8f;

// ---- WMMA fragment loaders (bf16 16x16x32, contiguous k order, same in A and B) ----

// A (16x32 of a row-major source): lane holds row m = lane&15; lanes<16 hold
// K 0..15, lanes>=16 hold K 16..31. Two contiguous 8-float
// spans -> four 16B vector loads, then packed f32->bf16 converts.
__device__ __forceinline__ v16bf load_a_frag(const float* rowptr, int k0, int lane) {
    int half = (lane >> 4) & 1;
    const float4* p0 = (const float4*)(rowptr + k0 + half * 16);       // K base+0..7
    const float4* p1 = (const float4*)(rowptr + k0 + half * 16 + 8);   // K base+8..15
    float4 x0 = p0[0], x1 = p0[1];
    float4 y0 = p1[0], y1 = p1[1];
    v16bf a;
    a[0]  = (__bf16)x0.x; a[1]  = (__bf16)x0.y; a[2]  = (__bf16)x0.z; a[3]  = (__bf16)x0.w;
    a[4]  = (__bf16)x1.x; a[5]  = (__bf16)x1.y; a[6]  = (__bf16)x1.z; a[7]  = (__bf16)x1.w;
    a[8]  = (__bf16)y0.x; a[9]  = (__bf16)y0.y; a[10] = (__bf16)y0.z; a[11] = (__bf16)y0.w;
    a[12] = (__bf16)y1.x; a[13] = (__bf16)y1.y; a[14] = (__bf16)y1.z; a[15] = (__bf16)y1.w;
    return a;
}

// B (32x16 block of row-major W[K x N]): lane col = lane&15; lanes<16 hold K 0..15,
// lanes>=16 hold K 16..31; VGPR j packs K=2j,2j+1 of that half. LDW is a
// compile-time constant so all 16 loads share one base address with immediate
// offsets (j*LDW*4 <= 61440 < 2^23), letting the scheduler batch them.
template <int LDW>
__device__ __forceinline__ v16bf load_b_frag(const float* W, int k0, int n0, int lane) {
    int half = (lane >> 4) & 1;
    int c = lane & 15;
    const float* base = W + (k0 + half * 16) * LDW + n0 + c;
    float tmp[16];
#pragma unroll
    for (int j = 0; j < 16; ++j) tmp[j] = base[j * LDW];
    v16bf b;
#pragma unroll
    for (int j = 0; j < 16; ++j) b[j] = (__bf16)tmp[j];
    return b;
}

// ---- Root MLP GEMM: one wave computes a 16x64 strip (4 WMMA tiles share A) ----
template <int N, int K, bool RELU>
__global__ __launch_bounds__(32) void root_gemm(const float* __restrict__ A,
                                                const float* __restrict__ W,
                                                const float* __restrict__ bias,
                                                float* __restrict__ C) {
    constexpr int NG = N >> 6;             // 64-wide n-groups
    int mt = blockIdx.x / NG;
    int ng = blockIdx.x % NG;
    int n0 = ng * 64;
    int lane = threadIdx.x;
    int half = (lane >> 4) & 1;
    int cc = lane & 15;
    const float* arow = A + (size_t)(mt * 16 + cc) * K;
    v8f acc[4] = {};
    for (int k0 = 0; k0 < K; k0 += 32) {
        v16bf a = load_a_frag(arow, k0, lane);
#pragma unroll
        for (int t = 0; t < 4; ++t) {
            v16bf b = load_b_frag<N>(W, k0, n0 + 16 * t, lane);
            acc[t] = __builtin_amdgcn_wmma_f32_16x16x32_bf16(false, a, false, b,
                                                             (short)0, acc[t], false, false);
        }
    }
#pragma unroll
    for (int t = 0; t < 4; ++t) {
        float bv = bias[n0 + 16 * t + cc];
#pragma unroll
        for (int r = 0; r < 8; ++r) {
            int m = mt * 16 + r + half * 8;
            float v = acc[t][r] + bv;
            if (RELU) v = fmaxf(v, 0.0f);
            C[(size_t)m * N + n0 + 16 * t + cc] = v;
        }
    }
}

// ---- Root reduce: argmax + log-softmax at argmax, one thread per row ----
__global__ void root_reduce(const float* __restrict__ logits, int* __restrict__ idx,
                            float* __restrict__ root_lp) {
    int b = blockIdx.x * blockDim.x + threadIdx.x;
    if (b >= BATCH) return;
    const float* r = logits + (size_t)b * EXP;
    float mx = r[0]; int am = 0;
    for (int j = 1; j < EXP; ++j) { float v = r[j]; if (v > mx) { mx = v; am = j; } }
    float s = 0.0f;
    for (int j = 0; j < EXP; ++j) s += expf(r[j] - mx);
    idx[b] = am;
    root_lp[b] = (r[am] - mx) - logf(s);
}

// ---- Dispatch: group sample ids by routed expert ----
__global__ void dispatch(const int* __restrict__ idx, int* __restrict__ counts,
                         int* __restrict__ lists) {
    int t = threadIdx.x;
    if (t < EXP) counts[t] = 0;
    __syncthreads();
    int e = idx[t];
    int slot = atomicAdd(&counts[e], 1);
    lists[e * BATCH + slot] = t;
}

// ---- Expert GEMM: Y[s,N] = act(X[s,K] @ eW[e] + eb[e]) for samples routed to e.
// Padded rows clamp to a valid sample row (branchless loads); their outputs are
// simply not written back.
template <int N, int K, bool RELU>
__global__ __launch_bounds__(32) void expert_gemm(const float* __restrict__ X,
                                                  const float* __restrict__ eW,
                                                  const float* __restrict__ eb,
                                                  float* __restrict__ Y,
                                                  const int* __restrict__ counts,
                                                  const int* __restrict__ lists) {
    constexpr int NG = N >> 6;
    int e   = blockIdx.x / (16 * NG);
    int rem = blockIdx.x % (16 * NG);
    int mt  = rem / NG;
    int ng  = rem % NG;
    int n0  = ng * 64;
    int count = counts[e];
    if (mt * 16 >= count) return;          // uniform early-exit for empty tiles
    int lane = threadIdx.x;
    int half = (lane >> 4) & 1;
    int cc = lane & 15;
    const int* lst = lists + e * BATCH;
    int ga = mt * 16 + cc;
    if (ga > count - 1) ga = count - 1;    // clamp: always a valid row
    const float* arow = X + (size_t)lst[ga] * K;
    const float* W = eW + (size_t)e * K * N;
    v8f acc[4] = {};
    for (int k0 = 0; k0 < K; k0 += 32) {
        v16bf a = load_a_frag(arow, k0, lane);
#pragma unroll
        for (int t = 0; t < 4; ++t) {
            v16bf b = load_b_frag<N>(W, k0, n0 + 16 * t, lane);
            acc[t] = __builtin_amdgcn_wmma_f32_16x16x32_bf16(false, a, false, b,
                                                             (short)0, acc[t], false, false);
        }
    }
    const float* bb = eb + (size_t)e * N + n0 + cc;
#pragma unroll
    for (int t = 0; t < 4; ++t) {
        float bv = bb[16 * t];
#pragma unroll
        for (int r = 0; r < 8; ++r) {
            int g = mt * 16 + r + half * 8;
            if (g < count) {
                float v = acc[t][r] + bv;
                if (RELU) v = fmaxf(v, 0.0f);
                Y[(size_t)lst[g] * N + n0 + 16 * t + cc] = v;
            }
        }
    }
}

// ---- Top-(K+1), drop rule, final outputs. One 256-thread block per sample ----
__global__ __launch_bounds__(256) void topk_out(const float* __restrict__ logits,
                                                const int* __restrict__ idx,
                                                const float* __restrict__ root_lp,
                                                float* __restrict__ out) {
    __shared__ float sv[NLEAF];
    __shared__ float red[256];
    __shared__ int   redi[256];
    __shared__ int   top_idx[KOUT + 1];
    int b = blockIdx.x;
    int t = threadIdx.x;
    const float* L = logits + (size_t)b * NLEAF;

    float mx = -1e30f;
    for (int j = t; j < NLEAF; j += 256) { float v = L[j]; sv[j] = v; mx = fmaxf(mx, v); }
    red[t] = mx; __syncthreads();
    for (int s = 128; s > 0; s >>= 1) { if (t < s) red[t] = fmaxf(red[t], red[t + s]); __syncthreads(); }
    float gmax = red[0]; __syncthreads();

    float ss = 0.0f;
    for (int j = t; j < NLEAF; j += 256) ss += expf(sv[j] - gmax);
    red[t] = ss; __syncthreads();
    for (int s = 128; s > 0; s >>= 1) { if (t < s) red[t] += red[t + s]; __syncthreads(); }
    float lse = gmax + logf(red[0]); __syncthreads();

    // iterative stable top-(K+1): ties resolved toward lower index (matches lax.top_k)
    for (int r = 0; r < KOUT + 1; ++r) {
        float bv = -1e30f; int bi = 0x7fffffff;
        for (int j = t; j < NLEAF; j += 256) {
            float v = sv[j];
            if (v > bv || (v == bv && j < bi)) { bv = v; bi = j; }
        }
        red[t] = bv; redi[t] = bi; __syncthreads();
        for (int s = 128; s > 0; s >>= 1) {
            if (t < s) {
                if (red[t + s] > red[t] || (red[t + s] == red[t] && redi[t + s] < redi[t])) {
                    red[t] = red[t + s]; redi[t] = redi[t + s];
                }
            }
            __syncthreads();
        }
        if (t == 0) { top_idx[r] = redi[0]; sv[redi[0]] = -1e30f; }
        __syncthreads();
    }

    if (t == 0) {
        int e = idx[b];
        float rlp = root_lp[b];
        float* traj = out;                               // [B, K, 2] as floats
        float* flp  = out + (size_t)BATCH * KOUT * 2;    // [B, K]
        int outn = 0;
        for (int r = 0; r < KOUT + 1 && outn < KOUT; ++r) {
            int it = top_idx[r];
            if (it == 0 && e == 0) continue;             // drop the (0,0) trajectory
            traj[((size_t)b * KOUT + outn) * 2 + 0] = (float)e;
            traj[((size_t)b * KOUT + outn) * 2 + 1] = (float)it;
            flp[(size_t)b * KOUT + outn] = (L[it] - lse) + rlp;
            ++outn;
        }
    }
}

extern "C" void kernel_launch(void* const* d_in, const int* in_sizes, int n_in,
                              void* d_out, int out_size, void* d_ws, size_t ws_size,
                              hipStream_t stream) {
    const float* state = (const float*)d_in[0];
    const float* rW1   = (const float*)d_in[1];
    const float* rb1   = (const float*)d_in[2];
    const float* rW2   = (const float*)d_in[3];
    const float* rb2   = (const float*)d_in[4];
    const float* rW3   = (const float*)d_in[5];
    const float* rb3   = (const float*)d_in[6];
    const float* eW1   = (const float*)d_in[7];
    const float* eb1   = (const float*)d_in[8];
    const float* eW2   = (const float*)d_in[9];
    const float* eb2   = (const float*)d_in[10];
    const float* eW3   = (const float*)d_in[11];
    const float* eb3   = (const float*)d_in[12];

    char* ws = (char*)d_ws;
    size_t o = 0;
    float* h1   = (float*)(ws + o); o += (size_t)BATCH * HID   * 4;   // 512KB
    float* h2   = (float*)(ws + o); o += (size_t)BATCH * HID   * 4;   // 512KB
    float* rlog = (float*)(ws + o); o += (size_t)BATCH * EXP   * 4;   // 64KB
    float* elog = (float*)(ws + o); o += (size_t)BATCH * NLEAF * 4;   // 1MB
    float* eh1  = (float*)(ws + o); o += (size_t)BATCH * HID   * 4;   // 512KB
    float* eh2  = (float*)(ws + o); o += (size_t)BATCH * HID   * 4;   // 512KB
    int*   idx  = (int*)  (ws + o); o += (size_t)BATCH * 4;
    float* rlp  = (float*)(ws + o); o += (size_t)BATCH * 4;
    int*   cnts = (int*)  (ws + o); o += (size_t)EXP * 4;
    int*   lsts = (int*)  (ws + o); o += (size_t)EXP * BATCH * 4;     // 64KB

    dim3 wv(32);
    // Root MLP (n-groups of 64 columns per wave)
    root_gemm<HID, DIN, true ><<<dim3((BATCH / 16) * (HID / 64)), wv, 0, stream>>>(state, rW1, rb1, h1);
    root_gemm<HID, HID, true ><<<dim3((BATCH / 16) * (HID / 64)), wv, 0, stream>>>(h1, rW2, rb2, h2);
    root_gemm<EXP, HID, false><<<dim3((BATCH / 16) * (EXP / 64)), wv, 0, stream>>>(h2, rW3, rb3, rlog);
    root_reduce<<<1, 256, 0, stream>>>(rlog, idx, rlp);
    dispatch<<<1, 256, 0, stream>>>(idx, cnts, lsts);
    // Expert MLP, samples grouped by expert so each expert's weights stream once
    expert_gemm<HID,   DIN, true ><<<dim3(EXP * 16 * (HID / 64)),   wv, 0, stream>>>(state, eW1, eb1, eh1, cnts, lsts);
    expert_gemm<HID,   HID, true ><<<dim3(EXP * 16 * (HID / 64)),   wv, 0, stream>>>(eh1, eW2, eb2, eh2, cnts, lsts);
    expert_gemm<NLEAF, HID, false><<<dim3(EXP * 16 * (NLEAF / 64)), wv, 0, stream>>>(eh2, eW3, eb3, elog, cnts, lsts);
    // Top-K + drop rule + outputs
    topk_out<<<BATCH, 256, 0, stream>>>(elog, idx, rlp, (float*)d_out);
}
